// MultiheadRoPESelfAttention_3702261809282
// MI455X (gfx1250) — matrix operand order repaired and checked by reading, not compiled, not run on hardware
//
#include <hip/hip_runtime.h>
#include <hip/hip_bf16.h>

typedef __attribute__((ext_vector_type(16))) _Float16 v16h;
typedef __attribute__((ext_vector_type(8)))  _Float16 v8h;
typedef __attribute__((ext_vector_type(4)))  _Float16 v4h;
typedef __attribute__((ext_vector_type(8)))  float    v8f;

namespace {

constexpr int kB  = 4;
constexpr int kT  = 2048;
constexpr int kD  = 1024;
constexpr int kH  = 16;
constexpr int kDh = 64;

__device__ inline v8f zero8() {
  v8f z = {0.f, 0.f, 0.f, 0.f, 0.f, 0.f, 0.f, 0.f};
  return z;
}

// ---- WMMA fragment loaders (layouts per CDNA5 ISA 7.12.2), f16 sources ----
// A fragment 16x32 f16: lane m = lane&15, lh = lane>>4.
//   e in 0..7  -> K = kBase + 8*lh + e
//   e in 8..15 -> K = kBase + 8*lh + 16 + (e-8)
// Two aligned 16-byte vector loads -> global_load_b128 each.
__device__ inline v16h a_frag_f16(const _Float16* __restrict__ src, int ld,
                                  int mBase, int kBase, int lane) {
  const int m = lane & 15, lh = lane >> 4;
  const _Float16* p = src + (size_t)(mBase + m) * ld + kBase + 8 * lh;
  const v8h lo = *(const v8h*)p;
  const v8h hi = *(const v8h*)(p + 16);
  v16h a;
#pragma unroll
  for (int e = 0; e < 8; ++e) { a[e] = lo[e]; a[8 + e] = hi[e]; }
  return a;
}

// B fragment 32x16: element (k, n) = src[(nBase+n)*ld + kBase + k]
//   lane n = lane&15, lh = lane>>4; e in 0..7 -> k = 8*lh + e,
//   e in 8..15 -> k = 8*lh + 16 + (e-8) (two 16-byte v8h loads).
__device__ inline v16h b_frag_f16(const _Float16* __restrict__ src, int ld,
                                  int nBase, int kBase, int lane) {
  const int n = lane & 15, lh = lane >> 4;
  const _Float16* p = src + (size_t)(nBase + n) * ld + kBase + 8 * lh;
  const v8h lo = *(const v8h*)p;
  const v8h hi = *(const v8h*)(p + 16);
  v16h b;
#pragma unroll
  for (int e = 0; e < 8; ++e) { b[e] = lo[e]; b[8 + e] = hi[e]; }
  return b;
}

__device__ inline v8f wmma_f16(v16h a, v16h b, v8f c) {
  return __builtin_amdgcn_wmma_f32_16x16x32_f16(false, a, false, b, (short)0, c,
                                                false, false);
}

}  // namespace

// ============================================================================
// Kernel 0: one-time f32 -> f16 conversion (x, W_qkv, W_out) into workspace.
// ============================================================================
__global__ __launch_bounds__(256) void cvt_f32_f16_kernel(
    const float* __restrict__ src, _Float16* __restrict__ dst, int n) {
  const int i = (blockIdx.x * 256 + threadIdx.x) * 4;
  if (i < n) {
    const float4 v = *(const float4*)(src + i);
    v4h o;
    o[0] = (_Float16)v.x;
    o[1] = (_Float16)v.y;
    o[2] = (_Float16)v.z;
    o[3] = (_Float16)v.w;
    *(v4h*)(dst + i) = o;
  }
}

// ============================================================================
// Kernel 1: qkv = x @ W_qkv^T  (M=8192, N=3072, K=1024), fused RoPE epilogue.
// One wave -> 32x64 output tile (8 accumulators; 2 A-frags reuse 4 B-frags
// per K-step -> 8 WMMA / K-step). Writes f16 workspace:
//   q,k (post-RoPE): [b, h, t, 64] row-major;  v: transposed [b, h, 64, t].
// ============================================================================
__global__ __launch_bounds__(256) void qkv_rope_kernel(
    const _Float16* __restrict__ xh, const _Float16* __restrict__ Wqkvh,
    _Float16* __restrict__ qws, _Float16* __restrict__ kws,
    _Float16* __restrict__ vtws) {
  const int lane = threadIdx.x & 31;
  const int wave = blockIdx.x * (blockDim.x >> 5) + (threadIdx.x >> 5);
  constexpr int nGroups = (3 * kD) / 64;  // 48
  const int mBase = (wave / nGroups) * 32;
  const int nBase = (wave % nGroups) * 64;

  v8f acc[2][4];
#pragma unroll
  for (int mi = 0; mi < 2; ++mi)
#pragma unroll
    for (int j = 0; j < 4; ++j) acc[mi][j] = zero8();

  for (int kb = 0; kb < kD; kb += 32) {
    const v16h a0 = a_frag_f16(xh, kD, mBase, kb, lane);
    const v16h a1 = a_frag_f16(xh, kD, mBase + 16, kb, lane);
    if (kb + 32 < kD)  // hint next A row chunk toward the caches
      __builtin_prefetch(xh + (size_t)(mBase + (lane & 15)) * kD + kb + 32, 0, 1);
#pragma unroll
    for (int j = 0; j < 4; ++j) {
      const v16h bf = b_frag_f16(Wqkvh, kD, nBase + 16 * j, kb, lane);
      acc[0][j] = wmma_f16(a0, bf, acc[0][j]);
      acc[1][j] = wmma_f16(a1, bf, acc[1][j]);
    }
  }

  const int lh = lane >> 4;
  const int nl = lane & 15;
  const int bIdx = mBase / kT;  // 32-row tiles never cross batches

#pragma unroll
  for (int mi = 0; mi < 2; ++mi) {
    const int tBase = ((mBase + 16 * mi) % kT) + 8 * lh;
#pragma unroll
    for (int j = 0; j < 4; ++j) {
      const int n = nBase + 16 * j + nl;
      const int sect = n >> 10;  // 0=q, 1=k, 2=v  (uniform across the wave)
      const int d = n & 63;
      const int head = (n >> 6) & 15;
      if (sect < 2) {
        // RoPE: pair (2i,2i+1) sits in adjacent lanes (n parity==lane parity)
        const int i = d >> 1;
        const float freq = __powf(10000.0f, -(float)(2 * i) * (1.0f / 64.0f));
        const bool even = (d & 1) == 0;
        _Float16* dst = (sect == 0) ? qws : kws;
        _Float16* base = dst + ((size_t)(bIdx * kH + head) * kT) * kDh + d;
#pragma unroll
        for (int r = 0; r < 8; ++r) {
          const float v0 = acc[mi][j][r];
          const float pv = __shfl_xor(v0, 1, 32);
          const int t = tBase + r;
          const float ang = (float)t * freq;
          const float c = __cosf(ang);
          const float s = __sinf(ang);
          const float o = even ? (v0 * c - pv * s) : (pv * s + v0 * c);
          base[(size_t)t * kDh] = (_Float16)o;
        }
      } else {
        // V stored transposed: [b, h, d, t] -> contiguous attn B-fragments
        _Float16* base = vtws + ((size_t)(bIdx * kH + head) * kDh + d) * kT;
#pragma unroll
        for (int r = 0; r < 8; ++r) base[tBase + r] = (_Float16)acc[mi][j][r];
      }
    }
  }
}

// ============================================================================
// Kernel 2: causal flash attention. One wave = (b, h, 16-query tile),
// 32-key blocks: S = Q*K^T (4 WMMA), online softmax (16-lane shfl reductions),
// P restaged via per-wave LDS (D-frag -> A-frag), O += P*V (4 WMMA).
// Output f16 [b, t, 1024].
// ============================================================================
__global__ __launch_bounds__(256) void flash_attn_kernel(
    const _Float16* __restrict__ qws, const _Float16* __restrict__ kws,
    const _Float16* __restrict__ vtws, _Float16* __restrict__ ows) {
  __shared__ _Float16 pstage[8][16 * 32];  // 1KB per wave
  const int lane = threadIdx.x & 31;
  const int wslot = threadIdx.x >> 5;
  const int wave = blockIdx.x * 8 + wslot;
  constexpr int nQT = kT / 16;  // 128
  const int qt = wave % nQT;
  const int bh = wave / nQT;
  const int bIdx = bh / kH, head = bh % kH;

  const _Float16* Qp = qws + ((size_t)(bIdx * kH + head) * kT) * kDh;
  const _Float16* Kp = kws + ((size_t)(bIdx * kH + head) * kT) * kDh;
  const _Float16* Vt = vtws + ((size_t)(bIdx * kH + head) * kDh) * kT;
  const int qBase = qt * 16;
  const int nl = lane & 15, lh = lane >> 4;

  // Q fragments for the whole head dim (K = 0..63), loaded once
  const v16h qa0 = a_frag_f16(Qp, kDh, qBase, 0, lane);
  const v16h qa1 = a_frag_f16(Qp, kDh, qBase, 32, lane);

  v8f oacc[4];
#pragma unroll
  for (int j = 0; j < 4; ++j) oacc[j] = zero8();
  float rmax[8], rsum[8];
#pragma unroll
  for (int r = 0; r < 8; ++r) { rmax[r] = -1e30f; rsum[r] = 0.0f; }

  const float scale = 0.125f;  // 1/sqrt(64)
  _Float16* stage = &pstage[wslot][0];

  for (int kb = 0; kb < qBase + 16; kb += 32) {
    v8f s0 = zero8(), s1 = zero8();
    {
      const v16h k00 = b_frag_f16(Kp, kDh, kb, 0, lane);
      const v16h k01 = b_frag_f16(Kp, kDh, kb, 32, lane);
      s0 = wmma_f16(qa0, k00, s0);
      s0 = wmma_f16(qa1, k01, s0);
      const v16h k10 = b_frag_f16(Kp, kDh, kb + 16, 0, lane);
      const v16h k11 = b_frag_f16(Kp, kDh, kb + 16, 32, lane);
      s1 = wmma_f16(qa0, k10, s1);
      s1 = wmma_f16(qa1, k11, s1);
    }
#pragma unroll
    for (int r = 0; r < 8; ++r) {
      const int qg = qBase + r + 8 * lh;
      float a0 = s0[r] * scale;
      float a1 = s1[r] * scale;
      if (kb + nl > qg) a0 = -1e30f;  // causal mask
      if (kb + 16 + nl > qg) a1 = -1e30f;
      float bm = fmaxf(a0, a1);  // row max across the 16 lanes of this half
#pragma unroll
      for (int off = 1; off < 16; off <<= 1)
        bm = fmaxf(bm, __shfl_xor(bm, off, 32));
      const float mnew = fmaxf(rmax[r], bm);
      const float corr = __expf(rmax[r] - mnew);
      const float p0 = __expf(a0 - mnew);
      const float p1 = __expf(a1 - mnew);
      float bs = p0 + p1;
#pragma unroll
      for (int off = 1; off < 16; off <<= 1) bs += __shfl_xor(bs, off, 32);
      rsum[r] = rsum[r] * corr + bs;
      rmax[r] = mnew;
#pragma unroll
      for (int j = 0; j < 4; ++j) oacc[j][r] *= corr;
      // stage P tile row-major [query][key 0..31] for A-fragment reload
      stage[(r + 8 * lh) * 32 + nl] = (_Float16)p0;
      stage[(r + 8 * lh) * 32 + 16 + nl] = (_Float16)p1;
    }
    asm volatile("s_wait_dscnt 0" ::: "memory");  // LDS RAW within wave
    const v16h pa = a_frag_f16(stage, 32, 0, 0, lane);
#pragma unroll
    for (int j = 0; j < 4; ++j) {
      const v16h vb = b_frag_f16(Vt, kT, j * 16, kb, lane);  // Vt rows = d
      oacc[j] = wmma_f16(pa, vb, oacc[j]);
    }
    asm volatile("s_wait_dscnt 0" ::: "memory");  // WAR before next restage
  }

#pragma unroll
  for (int r = 0; r < 8; ++r) {
    const float inv = 1.0f / rsum[r];
    const int t = qBase + r + 8 * lh;
#pragma unroll
    for (int j = 0; j < 4; ++j) {
      const int d = j * 16 + nl;
      ows[(size_t)(bIdx * kT + t) * kD + head * kDh + d] =
          (_Float16)(oacc[j][r] * inv);
    }
  }
}

// ============================================================================
// Kernel 3: out = o @ W_out^T  (M=8192, N=1024, K=1024), 32x64 wave tiles,
// f32 output.
// ============================================================================
__global__ __launch_bounds__(256) void out_proj_kernel(
    const _Float16* __restrict__ ows, const _Float16* __restrict__ Wouth,
    float* __restrict__ out) {
  const int lane = threadIdx.x & 31;
  const int wave = blockIdx.x * (blockDim.x >> 5) + (threadIdx.x >> 5);
  constexpr int nGroups = kD / 64;  // 16
  const int mBase = (wave / nGroups) * 32;
  const int nBase = (wave % nGroups) * 64;

  v8f acc[2][4];
#pragma unroll
  for (int mi = 0; mi < 2; ++mi)
#pragma unroll
    for (int j = 0; j < 4; ++j) acc[mi][j] = zero8();

  for (int kb = 0; kb < kD; kb += 32) {
    const v16h a0 = a_frag_f16(ows, kD, mBase, kb, lane);
    const v16h a1 = a_frag_f16(ows, kD, mBase + 16, kb, lane);
    if (kb + 32 < kD)
      __builtin_prefetch(ows + (size_t)(mBase + (lane & 15)) * kD + kb + 32, 0, 1);
#pragma unroll
    for (int j = 0; j < 4; ++j) {
      const v16h bf = b_frag_f16(Wouth, kD, nBase + 16 * j, kb, lane);
      acc[0][j] = wmma_f16(a0, bf, acc[0][j]);
      acc[1][j] = wmma_f16(a1, bf, acc[1][j]);
    }
  }

  const int lh = lane >> 4, nl = lane & 15;
#pragma unroll
  for (int mi = 0; mi < 2; ++mi)
#pragma unroll
    for (int j = 0; j < 4; ++j)
#pragma unroll
      for (int r = 0; r < 8; ++r)
        out[(size_t)(mBase + 16 * mi + r + 8 * lh) * kD + nBase + 16 * j + nl] =
            acc[mi][j][r];
}

extern "C" void kernel_launch(void* const* d_in, const int* in_sizes, int n_in,
                              void* d_out, int out_size, void* d_ws,
                              size_t ws_size, hipStream_t stream) {
  (void)in_sizes; (void)n_in; (void)out_size; (void)ws_size;
  const float* x = (const float*)d_in[0];
  const float* Wqkv = (const float*)d_in[1];
  const float* Wout = (const float*)d_in[2];
  float* out = (float*)d_out;

  // Workspace layout (halfs):
  //   q, k, vT, o : 4 x B*H*T*Dh = 4 x 8,388,608
  //   xh          : B*T*D       = 8,388,608
  //   Wqkvh       : 3*D*D       = 3,145,728
  //   Wouth       : D*D         = 1,048,576     (total ~88 MB)
  const size_t per = (size_t)kB * kH * kT * kDh;
  _Float16* qws = (_Float16*)d_ws;
  _Float16* kws = qws + per;
  _Float16* vtws = kws + per;
  _Float16* ows = vtws + per;
  _Float16* xh = ows + per;
  _Float16* Wqkvh = xh + (size_t)kB * kT * kD;
  _Float16* Wouth = Wqkvh + (size_t)3 * kD * kD;

  const int nX = kB * kT * kD;       // 8,388,608
  const int nWq = 3 * kD * kD;       // 3,145,728
  const int nWo = kD * kD;           // 1,048,576
  cvt_f32_f16_kernel<<<nX / 1024, 256, 0, stream>>>(x, xh, nX);
  cvt_f32_f16_kernel<<<nWq / 1024, 256, 0, stream>>>(Wqkv, Wqkvh, nWq);
  cvt_f32_f16_kernel<<<nWo / 1024, 256, 0, stream>>>(Wout, Wouth, nWo);

  {
    const int waves = (kB * kT / 32) * ((3 * kD) / 64);  // 12288
    qkv_rope_kernel<<<waves / 8, 256, 0, stream>>>(xh, Wqkvh, qws, kws, vtws);
  }
  {
    const int waves = kB * kH * (kT / 16);  // 8192
    flash_attn_kernel<<<waves / 8, 256, 0, stream>>>(qws, kws, vtws, ows);
  }
  {
    const int waves = (kB * kT / 32) * (kD / 64);  // 4096
    out_proj_kernel<<<waves / 8, 256, 0, stream>>>(ows, Wouth, out);
  }
}
